// EarthAttention3D_26233660244003
// MI455X (gfx1250) — matrix operand order repaired and checked by reading, not compiled, not run on hardware
//
#include <hip/hip_runtime.h>
#include <hip/hip_bf16.h>

typedef _Float16 half_t;
typedef __attribute__((ext_vector_type(16))) _Float16 v16h;
typedef __attribute__((ext_vector_type(8)))  _Float16 v8h;
typedef __attribute__((ext_vector_type(8)))  float    v8f;

#define NUM_HEADS 6
#define HD   32
#define NWIN 64
#define NTOK 144
#define CDIM 192
#define NPAD 160   // K-padded attention length (144 -> 160 = 5*32)

// ---- LDS layout (units: half elements); every offset is a multiple of 16 halves ----
#define XS_OFF  0                      // x tile 144x192
#define QH_OFF  (XS_OFF + NTOK*CDIM)   // q 144x32 (row-major)
#define KH_OFF  (QH_OFF + NTOK*HD)     // k 144x32 (row-major)
#define VT_OFF  (KH_OFF + NTOK*HD)     // v TRANSPOSED 32x160 (cols 144..159 zero)
#define SC_OFF  (VT_OFF + HD*NPAD)     // scores/probs 144x160 (cols 144..159 zero)
#define OUT_OFF (SC_OFF + NTOK*NPAD)   // head-concat out 144x192
#define LDS_HALFS (OUT_OFF + NTOK*CDIM)

// ---- WMMA operand loaders (contiguous k order, same in A and B) ----
// A tile 16x32 f16 from row-major matrix; per-lane: 16 contiguous halves (K = 16*(lane>>4) ..).
__device__ __forceinline__ v16h load_a_rm(const half_t* base, int ld) {
  const int lane = threadIdx.x & 31;
  const half_t* p = base + (lane & 15) * ld + ((lane >> 4) << 4);
  const v8h lo = *(const v8h*)p;          // K = kb .. kb+7
  const v8h hi = *(const v8h*)(p + 8);    // K = kb+8 .. kb+15
  return __builtin_shufflevector(lo, hi, 0,1,2,3,4,5,6,7,8,9,10,11,12,13,14,15);
}

// B tile 32x16 where B(k,n) = base[n*ld + k]; per-lane 16 CONTIGUOUS halves.
__device__ __forceinline__ v16h load_b_ctg(const half_t* base, int ld) {
  const int lane = threadIdx.x & 31;
  return *(const v16h*)(base + (lane & 15) * ld + ((lane >> 4) << 4));
}

// B tile from pre-packed weight buffer: lane l owns halves [l*16, l*16+16).
__device__ __forceinline__ v16h load_b_pk(const half_t* tile) {
  return *(const v16h*)(tile + (threadIdx.x & 31) * 16);
}

__device__ __forceinline__ v8f wmma16(v16h a, v16h b, v8f c) {
  return __builtin_amdgcn_wmma_f32_16x16x32_f16(
      /*neg_a=*/false, a, /*neg_b=*/false, b,
      /*c_mod=*/(short)0, c, /*reuse_a=*/false, /*reuse_b=*/false);
}

// ---------------- weight repack: f32 row-major -> f16 WMMA-B tiles ----------------
// Tile (kt,nt) of a KxN matrix stored at tIdx = nt*(K/32)+kt, 512 halves per tile:
// element (kt*32 + (lane>>4)*16 + j, nt*16 + (lane&15)) -> tile[lane*16 + j].
__global__ void __launch_bounds__(256)
pack_b_tiles(const float* __restrict__ W, half_t* __restrict__ P, int N) {
  const int idx = blockIdx.x * 256 + threadIdx.x;           // one packed element
  const int j    = idx & 15;
  const int lane = (idx >> 4) & 31;
  const int t    = idx >> 9;
  const int kt   = t % (CDIM / 32);
  const int nt   = t / (CDIM / 32);
  const int row  = kt * 32 + ((lane >> 4) << 4) + j;
  const int col  = nt * 16 + (lane & 15);
  P[idx] = (half_t)W[row * N + col];
}

// ---------------- fused windowed attention ----------------
__global__ void __launch_bounds__(256)
earth_attn_kernel(const float* __restrict__ x,
                  const float* __restrict__ mask,
                  const float* __restrict__ bqkv,
                  const float* __restrict__ posb,
                  const float* __restrict__ bproj,
                  const half_t* __restrict__ wq_pk,
                  const half_t* __restrict__ wp_pk,
                  float* __restrict__ out)
{
  extern __shared__ half_t lds[];
  const int tid  = threadIdx.x;
  const int lane = tid & 31;
  const int wv   = tid >> 5;            // wave id 0..7
  const int bw   = blockIdx.x;          // b*64 + w
  const int w    = bw % NWIN;

  half_t* xs = lds + XS_OFF;
  half_t* qh = lds + QH_OFF;
  half_t* kh = lds + KH_OFF;
  half_t* vt = lds + VT_OFF;
  half_t* sc = lds + SC_OFF;
  half_t* os = lds + OUT_OFF;

  const int cn = lane & 15;             // C/D tile column
  const int cm = (lane >> 4) << 3;      // C/D tile row base

  // ---- stage 0: x tile -> LDS f16, zero K-pad regions once ----
  {
    const float* xsrc = x + (size_t)bw * NTOK * CDIM;
    for (int i = tid; i < NTOK * CDIM; i += 256) xs[i] = (half_t)xsrc[i];
    for (int i = tid; i < HD * 16;   i += 256) vt[(i >> 4) * NPAD + 144 + (i & 15)] = (half_t)0.f;
    for (int i = tid; i < NTOK * 16; i += 256) sc[(i / 16) * NPAD + 144 + (i % 16)] = (half_t)0.f;
  }
  __syncthreads();

  const float scale = 0.17677669529663687f;   // 32^-0.5

  for (int h = 0; h < NUM_HEADS; ++h) {
    // ---- stage A: q,k,v (each 144x32) = xs(144x192) @ Wqkv head slice ----
    for (int t = wv; t < 54; t += 8) {          // 9 Mtiles x 3 mats x 2 Ntiles
      const int mt  = t / 6;
      const int r   = t % 6;
      const int mat = r >> 1;                   // 0=q 1=k 2=v
      const int nt  = r & 1;
      const int ntg = mat * (CDIM / 16) + h * 2 + nt;   // global N-tile in packed wqkv
      const half_t* btile = wq_pk + ((size_t)ntg * 6) * 512;
      v8f acc = {};
#pragma unroll
      for (int kk = 0; kk < 6; ++kk) {
        v16h a  = load_a_rm(xs + mt * 16 * CDIM + kk * 32, CDIM);
        v16h bb = load_b_pk(btile + kk * 512);
        acc = wmma16(a, bb, acc);
      }
      const float bias = bqkv[mat * CDIM + h * HD + nt * 16 + cn];
      if (mat == 2) {
        v8h pk;
#pragma unroll
        for (int i = 0; i < 8; ++i) pk[i] = (half_t)(acc[i] + bias);
        // transposed store: row n = nt*16+cn of vt, k = mt*16+cm.. contiguous
        *(v8h*)(vt + (nt * 16 + cn) * NPAD + mt * 16 + cm) = pk;
      } else {
        half_t* dst = (mat == 0) ? qh : kh;
#pragma unroll
        for (int i = 0; i < 8; ++i) {
          float v = acc[i] + bias;
          if (mat == 0) v *= scale;             // scale folded into q
          dst[(mt * 16 + cm + i) * HD + nt * 16 + cn] = (half_t)v;
        }
      }
    }
    __syncthreads();

    // ---- stage B: scores = q @ k^T + pos_bias + mask  (f16 into padded buffer) ----
    const float* pb = posb + (((size_t)h * NWIN + w) * NTOK) * NTOK;
    const float* mk = mask + ((size_t)bw * NTOK) * NTOK;
    for (int t = wv; t < 81; t += 8) {          // 9x9 tiles, K=32 single step
      const int mt = t / 9, nt = t % 9;
      v16h a  = load_a_rm(qh + mt * 16 * HD, HD);
      v16h bb = load_b_ctg(kh + (nt * 16) * HD, HD);   // B(k,n) = K[n][k]
      v8f acc = {};
      acc = wmma16(a, bb, acc);
#pragma unroll
      for (int i = 0; i < 8; ++i) {
        const int row = mt * 16 + cm + i;
        const int col = nt * 16 + cn;
        const float v = acc[i] + pb[row * NTOK + col] + mk[row * NTOK + col];
        sc[row * NPAD + col] = (half_t)v;
      }
    }
    __syncthreads();

    // ---- stage C: row softmax (f32 math, in-place f16 probs) ----
    if (tid < NTOK) {
      half_t* rowp = sc + tid * NPAD;
      float mx = -1e30f;
      for (int c = 0; c < NTOK; ++c) { const float v = (float)rowp[c]; mx = v > mx ? v : mx; }
      float sum = 0.f;
      for (int c = 0; c < NTOK; ++c) sum += __expf((float)rowp[c] - mx);
      const float inv = 1.f / sum;
      for (int c = 0; c < NTOK; ++c) rowp[c] = (half_t)(__expf((float)rowp[c] - mx) * inv);
    }
    __syncthreads();

    // ---- stage D: head out = P(144x160) @ V(160x32) via transposed V ----
    for (int t = wv; t < 18; t += 8) {          // 9 Mtiles x 2 Ntiles, K=160 = 5 steps
      const int mt = t >> 1, nt = t & 1;
      v8f acc = {};
#pragma unroll
      for (int kk = 0; kk < 5; ++kk) {
        v16h a  = load_a_rm(sc + mt * 16 * NPAD + kk * 32, NPAD);
        v16h bb = load_b_ctg(vt + (nt * 16) * NPAD + kk * 32, NPAD);  // B(k,n)=vt[n][k]
        acc = wmma16(a, bb, acc);
      }
#pragma unroll
      for (int i = 0; i < 8; ++i)
        os[(mt * 16 + cm + i) * CDIM + h * HD + nt * 16 + cn] = (half_t)acc[i];
    }
    __syncthreads();                            // protect qh/kh/vt/sc for next head
  }

  // ---- stage E: proj  out = os(144x192) @ Wproj(192x192) + bproj  (f32 out) ----
  float* dst = out + (size_t)bw * NTOK * CDIM;
  for (int t = wv; t < 108; t += 8) {           // 9 Mtiles x 12 Ntiles, K=6 steps
    const int mt = t / 12, nt = t % 12;
    const half_t* btile = wp_pk + ((size_t)nt * 6) * 512;
    v8f acc = {};
#pragma unroll
    for (int kk = 0; kk < 6; ++kk) {
      v16h a  = load_a_rm(os + mt * 16 * CDIM + kk * 32, CDIM);
      v16h bb = load_b_pk(btile + kk * 512);
      acc = wmma16(a, bb, acc);
    }
    const float bias = bproj[nt * 16 + cn];
#pragma unroll
    for (int i = 0; i < 8; ++i)
      dst[(mt * 16 + cm + i) * CDIM + nt * 16 + cn] = acc[i] + bias;
  }
}

extern "C" void kernel_launch(void* const* d_in, const int* in_sizes, int n_in,
                              void* d_out, int out_size, void* d_ws, size_t ws_size,
                              hipStream_t stream) {
  (void)in_sizes; (void)n_in; (void)out_size; (void)ws_size;
  const float* x     = (const float*)d_in[0];   // (15,64,144,192)
  const float* mask  = (const float*)d_in[1];   // (15,64,144,144)
  const float* wqkv  = (const float*)d_in[2];   // (192,576)
  const float* bqkv  = (const float*)d_in[3];   // (576,)
  const float* posb  = (const float*)d_in[4];   // (1,6,64,144,144)
  const float* wproj = (const float*)d_in[5];   // (192,192)
  const float* bproj = (const float*)d_in[6];   // (192,)
  float* out = (float*)d_out;

  half_t* wq_pk = (half_t*)d_ws;                // 192*576 f16, B-tile packed
  half_t* wp_pk = wq_pk + CDIM * 3 * CDIM;      // 192*192 f16, B-tile packed

  pack_b_tiles<<<(CDIM * 3 * CDIM) / 256, 256, 0, stream>>>(wqkv, wq_pk, 3 * CDIM);
  pack_b_tiles<<<(CDIM * CDIM) / 256, 256, 0, stream>>>(wproj, wp_pk, CDIM);

  const size_t lds_bytes = (size_t)LDS_HALFS * sizeof(half_t);  // ~181 KB
  earth_attn_kernel<<<15 * NWIN, 256, lds_bytes, stream>>>(
      x, mask, bqkv, posb, bproj, wq_pk, wp_pk, out);
}
